// Attention_msa_15908558864818
// MI455X (gfx1250) — compile-verified, matrix operand order checked by reading, not run on hardware
//
#include <hip/hip_runtime.h>

// ---------------------------------------------------------------------------
// MI455X (gfx1250) fused masked dual-softmax attention.
//   A: QKV GEMM (bf16 WMMA, f32 acc) with fused per-head L2 norm epilogue
//   B: per-16-row workgroup, all 8 heads: QK^T (cls+reg) -> masked scaled
//      softmax in 320KB LDS -> averaged attn -> AV WMMA -> head-mean sim
//   C: raw = VnVn^T (K=1024 WMMA) fused with sim softmax + mask + renorm
// ---------------------------------------------------------------------------

typedef __attribute__((ext_vector_type(16))) __bf16 bf16x16;
typedef __attribute__((ext_vector_type(8)))  __bf16 bf16x8;
typedef __attribute__((ext_vector_type(8)))  float  f32x8;

#define N_TOK   2048
#define CDIM    1024
#define C3      3072
#define HEADS   8
#define DHEAD   128
#define SC      25.0f
#define SIM_TH  0.75f

// 16-bit 16x32 fragment layout (ISA 7.12.2): lane (m = lane&15, hi = lane>>4)
// holds K = 8*hi..8*hi+7 in VGPR0-3 and K = 16+8*hi..23+8*hi in VGPR4-7, i.e.
// two contiguous 8-element (16B) runs -> two 128-bit loads per fragment.
__device__ __forceinline__ bf16x16 load_frag_bf16(const __bf16* base, int stride, int lane) {
  int m = lane & 15, hi = lane >> 4;
  const __bf16* row = base + (size_t)m * stride + 8 * hi;
  bf16x8 lo = *(const bf16x8*)(row);        // K = 8hi .. 8hi+7
  bf16x8 up = *(const bf16x8*)(row + 16);   // K = 16+8hi .. 23+8hi
  bf16x16 f;
#pragma unroll
  for (int e = 0; e < 8; ++e) { f[e] = lo[e]; f[8 + e] = up[e]; }
  return f;
}

// Same runs from an f32 source (LDS attn plane / global W), packed to bf16.
__device__ __forceinline__ bf16x16 load_frag_f32(const float* base, int stride, int lane) {
  int m = lane & 15, hi = lane >> 4;
  const float* row = base + (size_t)m * stride + 8 * hi;
  float4 a0 = *(const float4*)(row);
  float4 a1 = *(const float4*)(row + 4);
  float4 b0 = *(const float4*)(row + 16);
  float4 b1 = *(const float4*)(row + 20);
  bf16x16 f;
  f[0] = (__bf16)a0.x; f[1] = (__bf16)a0.y; f[2]  = (__bf16)a0.z; f[3]  = (__bf16)a0.w;
  f[4] = (__bf16)a1.x; f[5] = (__bf16)a1.y; f[6]  = (__bf16)a1.z; f[7]  = (__bf16)a1.w;
  f[8] = (__bf16)b0.x; f[9] = (__bf16)b0.y; f[10] = (__bf16)b0.z; f[11] = (__bf16)b0.w;
  f[12] = (__bf16)b1.x; f[13] = (__bf16)b1.y; f[14] = (__bf16)b1.z; f[15] = (__bf16)b1.w;
  return f;
}

__device__ __forceinline__ f32x8 wmma_bf16(bf16x16 a, bf16x16 b, f32x8 c) {
  return __builtin_amdgcn_wmma_f32_16x16x32_bf16(false, a, false, b, (short)0, c, false, false);
}

// ---------------------------------------------------------------------------
// Kernel A: qkv = x @ W^T, tile = 16 rows x one 128-wide (part,head) segment,
// fused L2 norm over d=128. 8 waves, one 16x16 WMMA tile each, K=1024.
// ---------------------------------------------------------------------------
__global__ __launch_bounds__(256) void qkv_kernel(
    const float* __restrict__ x_cls, const float* __restrict__ x_reg,
    const float* __restrict__ W_cls, const float* __restrict__ W_reg,
    __bf16* __restrict__ qkvn_cls, __bf16* __restrict__ qkvn_reg,
    __bf16* __restrict__ vT, float* __restrict__ out_x)
{
  __shared__ __bf16 aL[16 * CDIM];   // 32KB staged activations (bf16)
  __shared__ float  tile[16 * 128];  // 8KB raw GEMM result
  __shared__ float  red[16 * 16];
  __shared__ float  rnorm[16];

  const int t    = threadIdx.x;
  const int i0   = blockIdx.x * 16;
  const int seg  = blockIdx.y;            // part*8 + h
  const int part = seg >> 3, h = seg & 7;
  const int col0 = seg * 128;
  const int z    = blockIdx.z;            // 0=cls, 1=reg

  const float* X = z ? x_reg : x_cls;
  const float* W = z ? W_reg : W_cls;
  __bf16* qkvn   = z ? qkvn_reg : qkvn_cls;

  for (int idx = t; idx < 16 * CDIM; idx += 256) {
    int r = idx >> 10, c = idx & (CDIM - 1);
    aL[idx] = (__bf16)X[(size_t)(i0 + r) * CDIM + c];
  }
  __syncthreads();

  const int wv = t >> 5, lane = t & 31;
  const int hi = lane >> 4, cc16 = lane & 15;
  f32x8 acc = {};
  const float* Wbase = W + (size_t)(col0 + wv * 16) * CDIM;  // rows = out cols
#pragma unroll 4
  for (int kk = 0; kk < CDIM; kk += 32) {
    bf16x16 a = load_frag_bf16(aL + kk, CDIM, lane);
    bf16x16 b = load_frag_f32(Wbase + kk, CDIM, lane);
    acc = wmma_bf16(a, b, acc);
  }
#pragma unroll
  for (int v = 0; v < 8; ++v)
    tile[(v + 8 * hi) * 128 + wv * 16 + cc16] = acc[v];
  __syncthreads();

  // L2 norm over the 128-wide head segment
  {
    int r = t >> 4, sg = t & 15;
    float s = 0.f;
#pragma unroll
    for (int e = 0; e < 8; ++e) { float x = tile[r * 128 + sg * 8 + e]; s += x * x; }
    red[r * 16 + sg] = s;
  }
  __syncthreads();
  if ((t & 15) == 0) {
    int r = t >> 4; float s = 0.f;
#pragma unroll
    for (int e = 0; e < 16; ++e) s += red[r * 16 + e];
    rnorm[r] = rsqrtf(s);
  }
  __syncthreads();
  {
    int r = t >> 4, sg = t & 15;
    float rn = rnorm[r];
#pragma unroll
    for (int e = 0; e < 8; ++e) {
      int cc = sg * 8 + e;
      float val = tile[r * 128 + cc];
      qkvn[(size_t)(i0 + r) * C3 + col0 + cc] = (__bf16)(val * rn);  // normed q/k/v
      if (part == 2 && z == 0) {
        // raw v_cls: x_ori half of the output + bf16 v^T for the AV GEMM
        out_x[(size_t)(i0 + r) * (2 * CDIM) + CDIM + h * DHEAD + cc] = val;
        vT[(size_t)(h * DHEAD + cc) * N_TOK + (i0 + r)] = (__bf16)val;
      }
    }
  }
}

// ---------------------------------------------------------------------------
// Kernel B: one WG owns 16 query rows across ALL heads.
// LDS: 2 x (16 x 2048 f32) logit planes (cls/reg) + q tiles -> 265KB/WGP.
// ---------------------------------------------------------------------------
__global__ __launch_bounds__(256) void attn_kernel(
    const __bf16* __restrict__ qkvn_cls, const __bf16* __restrict__ qkvn_reg,
    const __bf16* __restrict__ vT,
    const float* __restrict__ cls_score, const float* __restrict__ fg_score,
    float* __restrict__ out_x, float* __restrict__ sim)
{
  extern __shared__ char smem[];
  float*  lgc  = (float*)smem;               // 16*2048 f32 (cls logits / attn)
  float*  lgr  = lgc + 16 * N_TOK;           // 16*2048 f32 (reg logits)
  __bf16* qc   = (__bf16*)(lgr + 16 * N_TOK);
  __bf16* qr   = qc + 16 * DHEAD;
  float*  red  = (float*)(qr + 16 * DHEAD);  // 256
  float*  stat = red + 256;                  // 64: [maxc|maxr|sumc|sumr] x16

  const int t = threadIdx.x, wv = t >> 5, lane = t & 31;
  const int i0 = blockIdx.x * 16;
  const int hi = lane >> 4, cc16 = lane & 15;
  const int rr = t >> 4, sg = t & 15;        // softmax mapping: row, 128-wide block

  float sic[8], sif[8];                      // query-row scores for the mask
#pragma unroll
  for (int v = 0; v < 8; ++v) {
    sic[v] = cls_score[i0 + v + 8 * hi];
    sif[v] = fg_score[i0 + v + 8 * hi];
  }

  for (int h = 0; h < HEADS; ++h) {
    for (int idx = t; idx < 16 * DHEAD; idx += 256) {
      int r = idx >> 7, c = idx & 127;
      qc[idx] = qkvn_cls[(size_t)(i0 + r) * C3 + h * DHEAD + c];
      qr[idx] = qkvn_reg[(size_t)(i0 + r) * C3 + h * DHEAD + c];
    }
    __syncthreads();

    // --- QK^T (cls + reg), masked scaled logits into LDS ---
    for (int jt = wv; jt < N_TOK / 16; jt += 8) {
      int j0 = jt * 16;
      f32x8 ac = {}, ar = {};
      const __bf16* kc = qkvn_cls + (size_t)j0 * C3 + CDIM + h * DHEAD;
      const __bf16* kr = qkvn_reg + (size_t)j0 * C3 + CDIM + h * DHEAD;
      __builtin_prefetch(kc + (size_t)cc16 * C3, 0, 1);   // global_prefetch_b8
      __builtin_prefetch(kr + (size_t)cc16 * C3, 0, 1);
#pragma unroll
      for (int kk = 0; kk < DHEAD; kk += 32) {
        bf16x16 a = load_frag_bf16(qc + kk, DHEAD, lane);
        bf16x16 b = load_frag_bf16(kc + kk, C3, lane);
        ac = wmma_bf16(a, b, ac);
        a = load_frag_bf16(qr + kk, DHEAD, lane);
        b = load_frag_bf16(kr + kk, C3, lane);
        ar = wmma_bf16(a, b, ar);
      }
      int jg = j0 + cc16;
      float sjc = cls_score[jg], sjf = fg_score[jg];
#pragma unroll
      for (int v = 0; v < 8; ++v) {
        int r = v + 8 * hi;
        float mc = (sjc > sic[v] - 0.1f) ? 1.f : 0.f;
        float mf = (sjf > sif[v] - 0.1f) ? 1.f : 0.f;
        lgc[r * N_TOK + jg] = ac[v] * SC * sjc * mc;   // mask multiplies logits
        lgr[r * N_TOK + jg] = ar[v] * SC * sjf * mf;
      }
    }
    __syncthreads();

    // --- softmax stats: max then sum-exp, cls and reg ---
    {
      float m0 = -3.4e38f, m1 = -3.4e38f;
      for (int j = sg * 128; j < sg * 128 + 128; ++j) m0 = fmaxf(m0, lgc[rr * N_TOK + j]);
      red[rr * 16 + sg] = m0; __syncthreads();
      if (sg == 0) { float m = red[rr * 16]; for (int e = 1; e < 16; ++e) m = fmaxf(m, red[rr * 16 + e]); stat[rr] = m; }
      __syncthreads();
      for (int j = sg * 128; j < sg * 128 + 128; ++j) m1 = fmaxf(m1, lgr[rr * N_TOK + j]);
      red[rr * 16 + sg] = m1; __syncthreads();
      if (sg == 0) { float m = red[rr * 16]; for (int e = 1; e < 16; ++e) m = fmaxf(m, red[rr * 16 + e]); stat[16 + rr] = m; }
      __syncthreads();
      float mc = stat[rr], mr = stat[16 + rr], s0 = 0.f, s1 = 0.f;
      for (int j = sg * 128; j < sg * 128 + 128; ++j) s0 += __expf(lgc[rr * N_TOK + j] - mc);
      red[rr * 16 + sg] = s0; __syncthreads();
      if (sg == 0) { float s = 0.f; for (int e = 0; e < 16; ++e) s += red[rr * 16 + e]; stat[32 + rr] = s; }
      __syncthreads();
      for (int j = sg * 128; j < sg * 128 + 128; ++j) s1 += __expf(lgr[rr * N_TOK + j] - mr);
      red[rr * 16 + sg] = s1; __syncthreads();
      if (sg == 0) { float s = 0.f; for (int e = 0; e < 16; ++e) s += red[rr * 16 + e]; stat[48 + rr] = s; }
      __syncthreads();
    }

    // --- averaged attn back into lgc; head-mean sim (exclusive rows -> no atomics) ---
    {
      float mc = stat[rr], mr = stat[16 + rr];
      float ic = 1.f / stat[32 + rr], ir = 1.f / stat[48 + rr];
      float* simrow = sim + (size_t)(i0 + rr) * N_TOK;
      for (int j = sg * 128; j < sg * 128 + 128; ++j) {
        float a = 0.5f * (__expf(lgc[rr * N_TOK + j] - mc) * ic +
                          __expf(lgr[rr * N_TOK + j] - mr) * ir);
        lgc[rr * N_TOK + j] = a;
        float prev = (h == 0) ? 0.f : simrow[j];
        simrow[j] = prev + a * 0.125f;
      }
    }
    __syncthreads();

    // --- AV: x[16 x 128] = attn[16 x 2048] @ v[2048 x 128], K=2048 WMMA ---
    {
      f32x8 acc = {};
      const __bf16* vb = vT + (size_t)(h * DHEAD + wv * 16) * N_TOK;
#pragma unroll 4
      for (int kk = 0; kk < N_TOK; kk += 32) {
        __builtin_prefetch(vb + (size_t)cc16 * N_TOK + kk + 128, 0, 1);
        bf16x16 a = load_frag_f32(lgc + kk, N_TOK, lane);
        bf16x16 b = load_frag_bf16(vb + kk, N_TOK, lane);
        acc = wmma_bf16(a, b, acc);
      }
#pragma unroll
      for (int v = 0; v < 8; ++v)
        out_x[(size_t)(i0 + v + 8 * hi) * (2 * CDIM) + h * DHEAD + wv * 16 + cc16] = acc[v];
    }
    __syncthreads();
  }
}

// ---------------------------------------------------------------------------
// Kernel C: raw = (Vn Vn^T)/H over concatenated heads (K=1024 WMMA), fused
// with sim softmax -> similarity mask -> renormalize. In place on d_out sim.
// ---------------------------------------------------------------------------
__global__ __launch_bounds__(256) void sim_kernel(
    const __bf16* __restrict__ qkvn_cls, float* __restrict__ sim)
{
  extern __shared__ char smem[];
  float*  p    = (float*)smem;              // 16*2048 masked probs
  __bf16* avn  = (__bf16*)(p + 16 * N_TOK); // 16*1024 normed-v rows
  float*  red  = (float*)(avn + 16 * CDIM); // 256
  float*  stat = red + 256;                 // 48: max | sumexp | masked-sum

  const int t = threadIdx.x, wv = t >> 5, lane = t & 31;
  const int i0 = blockIdx.x * 16;
  const int hi = lane >> 4, cc16 = lane & 15;
  const int rr = t >> 4, sg = t & 15;

  for (int idx = t; idx < 16 * CDIM; idx += 256) {
    int r = idx >> 10, c = idx & (CDIM - 1);
    avn[idx] = qkvn_cls[(size_t)(i0 + r) * C3 + 2 * CDIM + c];
  }
  __syncthreads();

  // sim softmax stats
  float m = -3.4e38f;
  for (int j = sg * 128; j < sg * 128 + 128; ++j) m = fmaxf(m, sim[(size_t)(i0 + rr) * N_TOK + j]);
  red[rr * 16 + sg] = m; __syncthreads();
  if (sg == 0) { float mm = red[rr * 16]; for (int e = 1; e < 16; ++e) mm = fmaxf(mm, red[rr * 16 + e]); stat[rr] = mm; }
  __syncthreads();
  float s = 0.f, mx = stat[rr];
  for (int j = sg * 128; j < sg * 128 + 128; ++j) s += __expf(sim[(size_t)(i0 + rr) * N_TOK + j] - mx);
  red[rr * 16 + sg] = s; __syncthreads();
  if (sg == 0) { float ss = 0.f; for (int e = 0; e < 16; ++e) ss += red[rr * 16 + e]; stat[16 + rr] = ss; }
  __syncthreads();

  // raw similarity tiles + masked probabilities
  for (int jt = wv; jt < N_TOK / 16; jt += 8) {
    int j0 = jt * 16;
    f32x8 acc = {};
    const __bf16* vb = qkvn_cls + (size_t)j0 * C3 + 2 * CDIM;
    __builtin_prefetch(vb + (size_t)cc16 * C3, 0, 1);
#pragma unroll 4
    for (int kk = 0; kk < CDIM; kk += 32) {
      bf16x16 a = load_frag_bf16(avn + kk, CDIM, lane);
      bf16x16 b = load_frag_bf16(vb + kk, C3, lane);
      acc = wmma_bf16(a, b, acc);
    }
    int jg = j0 + cc16;
#pragma unroll
    for (int v = 0; v < 8; ++v) {
      int r = v + 8 * hi;
      float raw = acc[v] * 0.125f;          // head-mean of per-head dots
      float val = 0.f;
      if (raw > SIM_TH)
        val = __expf(sim[(size_t)(i0 + r) * N_TOK + jg] - stat[r]) / stat[16 + r];
      p[r * N_TOK + jg] = val;
    }
  }
  __syncthreads();

  // masked row sums + renormalized write-back
  float s2 = 0.f;
  for (int j = sg * 128; j < sg * 128 + 128; ++j) s2 += p[rr * N_TOK + j];
  red[rr * 16 + sg] = s2; __syncthreads();
  if (sg == 0) { float ss = 0.f; for (int e = 0; e < 16; ++e) ss += red[rr * 16 + e]; stat[32 + rr] = ss; }
  __syncthreads();
  float inv = 1.f / stat[32 + rr];
  for (int j = sg * 128; j < sg * 128 + 128; ++j)
    sim[(size_t)(i0 + rr) * N_TOK + j] = p[rr * N_TOK + j] * inv;
}

// ---------------------------------------------------------------------------
extern "C" void kernel_launch(void* const* d_in, const int* in_sizes, int n_in,
                              void* d_out, int out_size, void* d_ws, size_t ws_size,
                              hipStream_t stream) {
  (void)in_sizes; (void)n_in; (void)out_size; (void)ws_size;
  const float* x_cls     = (const float*)d_in[0];
  const float* x_reg     = (const float*)d_in[1];
  const float* cls_score = (const float*)d_in[2];
  const float* fg_score  = (const float*)d_in[3];
  const float* W_cls     = (const float*)d_in[4];
  const float* W_reg     = (const float*)d_in[5];

  float* out   = (float*)d_out;
  float* out_x = out;                                  // (2048, 2048)
  float* sim   = out + (size_t)N_TOK * 2 * CDIM;       // (2048, 2048)

  // Workspace: normalized bf16 qkv (cls, reg) + bf16 v^T  (~29.4 MB)
  __bf16* qkvn_cls = (__bf16*)d_ws;
  __bf16* qkvn_reg = qkvn_cls + (size_t)N_TOK * C3;
  __bf16* vT       = qkvn_reg + (size_t)N_TOK * C3;

  qkv_kernel<<<dim3(N_TOK / 16, 24, 2), 256, 0, stream>>>(
      x_cls, x_reg, W_cls, W_reg, qkvn_cls, qkvn_reg, vT, out_x);

  size_t smemB = (size_t)2 * 16 * N_TOK * 4   // logit planes
               + (size_t)2 * 16 * DHEAD * 2   // q tiles
               + 256 * 4 + 64 * 4;            // reductions (~265.3 KB of 320 KB WGP LDS)
  attn_kernel<<<dim3(N_TOK / 16), 256, smemB, stream>>>(
      qkvn_cls, qkvn_reg, vT, cls_score, fg_score, out_x, sim);

  size_t smemC = (size_t)16 * N_TOK * 4 + (size_t)16 * CDIM * 2 + 256 * 4 + 64 * 4;
  sim_kernel<<<dim3(N_TOK / 16), 256, smemC, stream>>>(qkvn_cls, sim);
}
